// LSTMModel_57638461112868
// MI455X (gfx1250) — compile-verified, matrix operand order checked by reading, not run on hardware
//
#include <hip/hip_runtime.h>

#define BB 4096
#define TT 256
#define II 8
#define H1C 100
#define H2C 50

typedef __attribute__((ext_vector_type(16))) __bf16 v16bf;
typedef __attribute__((ext_vector_type(8)))  float  v8f;
typedef __attribute__((ext_vector_type(4)))  unsigned int uivec4;

union FragU { uivec4 q[2]; v16bf v; };

__device__ __forceinline__ unsigned short f2bf(float f) {
  unsigned int u = __float_as_uint(f);
  u += 0x7fffu + ((u >> 16) & 1u);          // round-to-nearest-even
  return (unsigned short)(u >> 16);
}
__device__ __forceinline__ float sigm(float x) { return 1.f / (1.f + __expf(-x)); }
__device__ __forceinline__ float tanh_f(float x) { float e = __expf(2.f * x); return 1.f - 2.f / (e + 1.f); }

// Load a 16-element bf16 fragment for lane: elements [off, off+8) and [off+16, off+24)
// of a K-contiguous row (two ds_load_b128).
__device__ __forceinline__ v16bf ldfrag(const unsigned short* rowp, int off) {
  FragU f;
  const uivec4* p = (const uivec4*)(rowp + off);
  f.q[0] = p[0];
  f.q[1] = p[2];
  return f.v;
}

// LDS carve (bytes), all 16B aligned
#define OFF_WHH1 0          // ushort [400][128]  102400
#define OFF_WIH2 102400     // ushort [208][128]   53248
#define OFF_WHH2 155648     // ushort [208][64]    26624
#define OFF_WIH1 182272     // float  [400][8]     12800
#define OFF_B1   195072     // float  [400]         1600
#define OFF_B2   196672     // float  [208]          832
#define OFF_G1   197504     // float  [16][400]    25600
#define OFF_G2   223104     // float  [16][208]    13312
#define OFF_H1   236416     // ushort [16][128]     4096
#define OFF_H2   240512     // ushort [16][64]      2048
#define OFF_C1   242560     // float  [16][100]     6400
#define OFF_C2   248960     // float  [16][50]      3200
#define OFF_H2F  252160     // float  [16][50]      3200
#define OFF_XS   255360     // float  [16][8]        512
#define LDS_BYTES 255872

__global__ __launch_bounds__(256) void lstm2_fused(
    const float* __restrict__ x,
    const float* __restrict__ Wih1, const float* __restrict__ Whh1,
    const float* __restrict__ bih1, const float* __restrict__ bhh1,
    const float* __restrict__ Wih2, const float* __restrict__ Whh2,
    const float* __restrict__ bih2, const float* __restrict__ bhh2,
    const float* __restrict__ Wd,   const float* __restrict__ bd,
    float* __restrict__ out)
{
  extern __shared__ __align__(16) unsigned char smem[];
  unsigned short* sWhh1 = (unsigned short*)(smem + OFF_WHH1);
  unsigned short* sWih2 = (unsigned short*)(smem + OFF_WIH2);
  unsigned short* sWhh2 = (unsigned short*)(smem + OFF_WHH2);
  float* sWih1 = (float*)(smem + OFF_WIH1);
  float* sB1   = (float*)(smem + OFF_B1);
  float* sB2   = (float*)(smem + OFF_B2);
  float* sG1   = (float*)(smem + OFF_G1);
  float* sG2   = (float*)(smem + OFF_G2);
  unsigned short* sH1 = (unsigned short*)(smem + OFF_H1);
  unsigned short* sH2 = (unsigned short*)(smem + OFF_H2);
  float* sC1  = (float*)(smem + OFF_C1);
  float* sC2  = (float*)(smem + OFF_C2);
  float* sH2f = (float*)(smem + OFF_H2F);
  float* sXs  = (float*)(smem + OFF_XS);

  const int tid  = threadIdx.x;
  const int lane = tid & 31;
  const int wave = tid >> 5;
  const int g    = lane >> 4;     // K-half select for WMMA fragments
  const int nloc = lane & 15;     // row (A: m) / col (B: n) within tile
  const int b0   = blockIdx.x * 16;

  // ---- stage weights into LDS (bf16, K padded to 32-multiples with zeros) ----
  for (int idx = tid; idx < 400 * 128; idx += 256) {
    int r = idx >> 7, k = idx & 127;
    sWhh1[idx] = (k < H1C) ? f2bf(Whh1[r * H1C + k]) : (unsigned short)0;
  }
  for (int idx = tid; idx < 208 * 128; idx += 256) {
    int r = idx >> 7, k = idx & 127;
    sWih2[idx] = (r < 4 * H2C && k < H1C) ? f2bf(Wih2[r * H1C + k]) : (unsigned short)0;
  }
  for (int idx = tid; idx < 208 * 64; idx += 256) {
    int r = idx >> 6, k = idx & 63;
    sWhh2[idx] = (r < 4 * H2C && k < H2C) ? f2bf(Whh2[r * H2C + k]) : (unsigned short)0;
  }
  for (int idx = tid; idx < 400 * 8; idx += 256) sWih1[idx] = Wih1[idx];
  for (int idx = tid; idx < 400; idx += 256) sB1[idx] = bih1[idx] + bhh1[idx];
  for (int idx = tid; idx < 208; idx += 256) sB2[idx] = (idx < 4 * H2C) ? (bih2[idx] + bhh2[idx]) : 0.f;
  // zero initial states (incl. K padding regions, which must stay zero)
  for (int idx = tid; idx < 16 * 128; idx += 256) sH1[idx] = 0;
  for (int idx = tid; idx < 16 * 64; idx += 256)  sH2[idx] = 0;
  for (int idx = tid; idx < 16 * H1C; idx += 256) sC1[idx] = 0.f;
  for (int idx = tid; idx < 16 * H2C; idx += 256) sC2[idx] = 0.f;
  __syncthreads();

  for (int t = 0; t < TT; ++t) {
    // stage x tile [16][8]
    if (tid < 128) {
      int m = tid >> 3, i = tid & 7;
      sXs[tid] = x[(size_t)(b0 + m) * (TT * II) + t * II + i];
    }
    __syncthreads();

    // ---- GEMM1: gates1[16][400] = bias + x@Wih1^T (fp32 VALU) + h1@Whh1^T (WMMA bf16) ----
    {
      const unsigned short* arow = sH1 + nloc * 128;
      for (int tile = wave; tile < 25; tile += 8) {
        int gcol = tile * 16 + nloc;
        const float* wr = sWih1 + gcol * 8;
        float bias = sB1[gcol];
        v8f c;
        #pragma unroll
        for (int r = 0; r < 8; ++r) {
          const float* xr = sXs + (r + 8 * g) * 8;
          float acc = bias;
          #pragma unroll
          for (int i2 = 0; i2 < 8; ++i2) acc = fmaf(xr[i2], wr[i2], acc);
          c[r] = acc;
        }
        const unsigned short* brow = sWhh1 + gcol * 128;
        #pragma unroll
        for (int ch = 0; ch < 4; ++ch) {
          v16bf a = ldfrag(arow, ch * 32 + 8 * g);
          v16bf b = ldfrag(brow, ch * 32 + 8 * g);
          c = __builtin_amdgcn_wmma_f32_16x16x32_bf16(false, a, false, b, (short)0, c, false, false);
        }
        #pragma unroll
        for (int r = 0; r < 8; ++r) sG1[(r + 8 * g) * 400 + gcol] = c[r];
      }
    }
    __syncthreads();

    // ---- Elementwise layer 1 (fp32 cell state) ----
    for (int j = tid; j < 16 * H1C; j += 256) {
      int m = j / H1C, h = j - m * H1C;
      const float* gr = sG1 + m * 400;
      float i_ = sigm(gr[h]);
      float f_ = sigm(gr[H1C + h]);
      float g_ = tanh_f(gr[2 * H1C + h]);
      float o_ = sigm(gr[3 * H1C + h]);
      float cc = fmaf(f_, sC1[j], i_ * g_);
      sC1[j] = cc;
      float hh = o_ * tanh_f(cc);
      sH1[m * 128 + h] = f2bf(hh);
    }
    __syncthreads();

    // ---- GEMM2: gates2[16][200] = bias2 + h1@Wih2^T (K=128) + h2@Whh2^T (K=64), all WMMA ----
    {
      const unsigned short* arow1 = sH1 + nloc * 128;
      const unsigned short* arow2 = sH2 + nloc * 64;
      for (int tile = wave; tile < 13; tile += 8) {
        int gcol = tile * 16 + nloc;
        float bias = sB2[gcol];
        v8f c;
        #pragma unroll
        for (int r = 0; r < 8; ++r) c[r] = bias;
        const unsigned short* brow1 = sWih2 + gcol * 128;
        #pragma unroll
        for (int ch = 0; ch < 4; ++ch) {
          v16bf a = ldfrag(arow1, ch * 32 + 8 * g);
          v16bf b = ldfrag(brow1, ch * 32 + 8 * g);
          c = __builtin_amdgcn_wmma_f32_16x16x32_bf16(false, a, false, b, (short)0, c, false, false);
        }
        const unsigned short* brow2 = sWhh2 + gcol * 64;
        #pragma unroll
        for (int ch = 0; ch < 2; ++ch) {
          v16bf a = ldfrag(arow2, ch * 32 + 8 * g);
          v16bf b = ldfrag(brow2, ch * 32 + 8 * g);
          c = __builtin_amdgcn_wmma_f32_16x16x32_bf16(false, a, false, b, (short)0, c, false, false);
        }
        #pragma unroll
        for (int r = 0; r < 8; ++r) sG2[(r + 8 * g) * 208 + gcol] = c[r];
      }
    }
    __syncthreads();

    // ---- Elementwise layer 2 ----
    for (int j = tid; j < 16 * H2C; j += 256) {
      int m = j / H2C, h = j - m * H2C;
      const float* gr = sG2 + m * 208;
      float i_ = sigm(gr[h]);
      float f_ = sigm(gr[H2C + h]);
      float g_ = tanh_f(gr[2 * H2C + h]);
      float o_ = sigm(gr[3 * H2C + h]);
      float cc = fmaf(f_, sC2[j], i_ * g_);
      sC2[j] = cc;
      float hh = o_ * tanh_f(cc);
      sH2[m * 64 + h] = f2bf(hh);
      sH2f[j] = hh;
    }
    __syncthreads();
  }

  // ---- Dense head: out[b] = h2 @ Wd^T + bd ----
  if (tid < 16) {
    float acc = bd[0];
    const float* hr = sH2f + tid * H2C;
    #pragma unroll 10
    for (int h = 0; h < H2C; ++h) acc = fmaf(hr[h], Wd[h], acc);
    out[b0 + tid] = acc;
  }
}

extern "C" void kernel_launch(void* const* d_in, const int* in_sizes, int n_in,
                              void* d_out, int out_size, void* d_ws, size_t ws_size,
                              hipStream_t stream) {
  const float* x    = (const float*)d_in[0];
  const float* Wih1 = (const float*)d_in[1];
  const float* Whh1 = (const float*)d_in[2];
  const float* bih1 = (const float*)d_in[3];
  const float* bhh1 = (const float*)d_in[4];
  const float* Wih2 = (const float*)d_in[5];
  const float* Whh2 = (const float*)d_in[6];
  const float* bih2 = (const float*)d_in[7];
  const float* bhh2 = (const float*)d_in[8];
  const float* Wd   = (const float*)d_in[9];
  const float* bd   = (const float*)d_in[10];
  float* out = (float*)d_out;

  dim3 grid(BB / 16), block(256);
  hipLaunchKernelGGL(lstm2_fused, grid, block, LDS_BYTES, stream,
                     x, Wih1, Whh1, bih1, bhh1, Wih2, Whh2, bih2, bhh2, Wd, bd, out);
}
